// FullNet_43671227465974
// MI455X (gfx1250) — matrix operand order repaired and checked by reading, not compiled, not run on hardware
//
#include <hip/hip_runtime.h>

// ---------------------------------------------------------------------------
// Types for CDNA5 WMMA (wave32): v_wmma_f32_16x16x32_bf16
// ---------------------------------------------------------------------------
typedef __bf16 bf16;
typedef __attribute__((ext_vector_type(16))) __bf16 v16bf;
typedef __attribute__((ext_vector_type(8)))  __bf16 bf16x8;   // 16B: b128 loads/stores
typedef __attribute__((ext_vector_type(8)))  float  v8f;

// ---------------------------------------------------------------------------
// fp32 -> bf16 conversion (grid-stride)
// ---------------------------------------------------------------------------
__global__ void f2bf_kernel(const float* __restrict__ in, bf16* __restrict__ out, long n) {
  long i = (long)blockIdx.x * blockDim.x + threadIdx.x;
  long stride = (long)gridDim.x * blockDim.x;
  for (; i < n; i += stride) out[i] = (bf16)in[i];
}

// ---------------------------------------------------------------------------
// Weight pack: fp32 (OC x Ktot) -> bf16 zero-padded (Mpad x Kpad)
// Removes ALL guards from the GEMM A-path: rows >= OC and cols >= Ktot are 0,
// so the K tail contributes exactly zero regardless of what B stages there.
// ---------------------------------------------------------------------------
__global__ void pack_wgt_kernel(const float* __restrict__ w, bf16* __restrict__ out,
                                int OC, int Ktot, int Kpad, long total) {
  long i = (long)blockIdx.x * blockDim.x + threadIdx.x;
  long stride = (long)gridDim.x * blockDim.x;
  for (; i < total; i += stride) {
    const int m = (int)(i / Kpad);
    const int k = (int)(i % Kpad);
    out[i] = (m < OC && k < Ktot) ? (bf16)w[(long)m * Ktot + k] : (bf16)0.0f;
  }
}

// ---------------------------------------------------------------------------
// Implicit-GEMM VALID conv, NCHW, square kernel (compile-time KW).
//   M = OC (padded to 64), N = Nb*OH*OW, K = C*KW*KW (padded to 32)
// Block: 256 threads = 8 waves tiling 64(M) x 128(N).
// Wave: 32x32 macro-tile = 2x2 WMMA tiles -> 4 accumulators,
//       2 A-frags + 2 B-frags (2 x ds_load_b128 each) ->
//       4 x v_wmma_f32_16x16x32_bf16 per K-chunk.
// LDS double-buffered: one barrier per chunk, staging overlaps WMMA.
// A-staging: 1 aligned global_load_b128 + 1 ds_store_b128, no guards.
// B-staging: incremental im2col pointer walk (no per-element divides).
// ---------------------------------------------------------------------------
#define BM 64
#define BN 128
#define BK 32
#define LDSP 40   // row stride 40 elems = 80B: 16B-aligned rows, 20-bank skew

template <int KW>
__global__ __launch_bounds__(256)
void conv_wmma_kernel(const bf16* __restrict__ in, const bf16* __restrict__ wgt,
                      const float* __restrict__ bias, void* __restrict__ outp,
                      int Nb, int C, int H, int W, int OC, int Kpad,
                      int OH, int OW, int relu, int out_bf16) {
  __shared__ bf16 As[2][BM][LDSP];  // weights tile, K-major rows
  __shared__ bf16 Bs[2][BN][LDSP];  // input-patch tile, pixel-major rows

  const int tid  = threadIdx.x;
  const int lane = tid & 31;
  const int wave = tid >> 5;
  const int wm   = wave >> 2;      // 0..1 : 32-row M macro-tile
  const int wn   = wave & 3;       // 0..3 : 32-col N macro-tile
  const int half = lane >> 4;      // lanes 0-15 vs 16-31
  const int row  = lane & 15;

  constexpr int KK = KW * KW;
  const int Npix = Nb * OH * OW;
  const int mBase = blockIdx.y * BM;
  const int nBase = blockIdx.x * BN;

  // ---- A-staging coords: row = tid&63, contiguous K-run of 8 ----
  const int aml = tid & 63;
  const int akb = (tid >> 6) * 8;          // 0,8,16,24
  const bf16* __restrict__ wrowA = wgt + (long)(mBase + aml) * Kpad;

  // ---- B-staging coords: pixel = tid&127, contiguous K-run of 16 ----
  const int pcol = tid & 127;
  const int bkb  = (tid >> 7) * 16;        // 0,16
  const int bpix = nBase + pcol;
  const int bpc  = (bpix < Npix) ? bpix : 0;
  const int bn   = bpc / (OH * OW);
  const int brem = bpc % (OH * OW);
  const int boy  = brem / OW;
  const int box  = brem % OW;
  const bf16* __restrict__ inpx = in + ((long)bn * C * H + boy) * W + box;

  // ---- cooperative stage of one K-chunk into LDS buffer s ----
  auto loadStage = [&](int kb, int s) {
    // A: one aligned 16B copy (weights pre-padded -> no guards)
    *(bf16x8*)&As[s][aml][akb] = *(const bf16x8*)&wrowA[kb + akb];

    // B: decode run start once, then incremental (ic,ky,kx) pointer walk
    const int k0 = kb + bkb;
    int ic = k0 / KK;                 // compile-time KK -> mul/shift
    const int r2 = k0 - ic * KK;
    int ky = r2 / KW;
    int kx = r2 - ky * KW;
    const bf16* p = inpx + ((long)ic * H + ky) * W + kx;

    union { bf16 e[16]; bf16x8 q[2]; } bt;
#pragma unroll
    for (int j = 0; j < 16; ++j) {
      bt.e[j] = *p;
      ++kx; ++p;
      const bool wxw = (kx == KW);
      kx = wxw ? 0 : kx;
      p += wxw ? (W - KW) : 0;
      const int kyn = ky + (wxw ? 1 : 0);
      const bool wyw = (kyn == KW);
      ky = wyw ? 0 : kyn;
      p += wyw ? (long)(H - KW) * W : 0;
    }
    *(bf16x8*)&Bs[s][pcol][bkb]     = bt.q[0];
    *(bf16x8*)&Bs[s][pcol][bkb + 8] = bt.q[1];
  };

  v8f acc[2][2] = {};

  loadStage(0, 0);
  __syncthreads();
  int cur = 0;
  for (int kb = 0; kb < Kpad; kb += BK) {
    // stage next chunk into the other buffer (overlaps WMMA below)
    if (kb + BK < Kpad) {
      __builtin_prefetch(&wrowA[kb + BK], 0, 0);   // global_prefetch_b8
      loadStage(kb + BK, cur ^ 1);
    }

    // ---- fragments per ISA 16-bit A(16x32)/B(32x16) layouts ----
    // A lane view: K in [half*8, +8) then [16+half*8, +8) -> 2x b128
    // B lane view: K in [half*8, +8) then [16+half*8, +8) -> 2x b128
    union { v16bf v; bf16x8 q[2]; } afrag[2], bfrag[2];
#pragma unroll
    for (int mt = 0; mt < 2; ++mt) {
      const int am = wm * 32 + mt * 16 + row;        // A: M = lane&15
      afrag[mt].q[0] = *(const bf16x8*)&As[cur][am][half * 8];
      afrag[mt].q[1] = *(const bf16x8*)&As[cur][am][16 + half * 8];
    }
#pragma unroll
    for (int nt = 0; nt < 2; ++nt) {
      const int bp = wn * 32 + nt * 16 + row;        // B: N = lane&15
      bfrag[nt].q[0] = *(const bf16x8*)&Bs[cur][bp][half * 8];
      bfrag[nt].q[1] = *(const bf16x8*)&Bs[cur][bp][16 + half * 8];
    }
#pragma unroll
    for (int mt = 0; mt < 2; ++mt)
#pragma unroll
      for (int nt = 0; nt < 2; ++nt)
        acc[mt][nt] = __builtin_amdgcn_wmma_f32_16x16x32_bf16(
            false, afrag[mt].v, false, bfrag[nt].v,
            (short)0, acc[mt][nt], false, false);

    __syncthreads();
    cur ^= 1;
  }

  // ---- epilogue: C/D layout — VGPR r: lanes0-15 M=r, lanes16-31 M=8+r ----
#pragma unroll
  for (int nt = 0; nt < 2; ++nt) {
    const int opix = nBase + wn * 32 + nt * 16 + row;
    if (opix >= Npix) continue;
    const int n0 = opix / (OH * OW);
    const int rr = opix % (OH * OW);
    const int oy = rr / OW;
    const int ox = rr % OW;
    const long obase = (((long)n0 * OC) * OH + oy) * OW + ox;
#pragma unroll
    for (int mt = 0; mt < 2; ++mt) {
#pragma unroll
      for (int r = 0; r < 8; ++r) {
        const int oc = mBase + wm * 32 + mt * 16 + r + (half ? 8 : 0);
        if (oc < OC) {
          float v = acc[mt][nt][r] + bias[oc];
          if (relu) v = fmaxf(v, 0.0f);
          const long oidx = obase + (long)oc * OH * OW;
          if (out_bf16) ((bf16*)outp)[oidx] = (bf16)v;
          else          ((float*)outp)[oidx] = v;
        }
      }
    }
  }
}

// ---------------------------------------------------------------------------
// Bicubic warp: builds the 15-channel (12 warped + disp + novelLoc) bf16 tensor
// ---------------------------------------------------------------------------
__device__ __forceinline__ void cubic_w(float t, float w[4]) {
  const float A = -0.75f;
  const float t1 = t + 1.0f, t2 = 1.0f - t, t3 = 2.0f - t;
  w[0] = ((A * t1 - 5.0f * A) * t1 + 8.0f * A) * t1 - 4.0f * A;  // k2(t+1)
  w[1] = ((A + 2.0f) * t - (A + 3.0f)) * t * t + 1.0f;           // k1(t)
  w[2] = ((A + 2.0f) * t2 - (A + 3.0f)) * t2 * t2 + 1.0f;        // k1(1-t)
  w[3] = ((A * t3 - 5.0f * A) * t3 + 8.0f * A) * t3 - 4.0f * A;  // k2(2-t)
}

__global__ void warp_kernel(const float* __restrict__ cf, const float* __restrict__ disp,
                            bf16* __restrict__ out15, int Nb, int H, int W) {
  const int idx = blockIdx.x * blockDim.x + threadIdx.x;
  const int total = Nb * H * W;
  if (idx >= total) return;
  const int n = idx / (H * W);
  const int rem = idx % (H * W);
  const int u = rem / W;
  const int v = rem % W;
  const long chw = (long)H * W;

  const float d  = disp[(long)n * chw + rem];
  const float q0 = cf[((long)n * 22 + 20) * chw + rem];
  const float q1 = cf[((long)n * 22 + 21) * chw + rem];

#pragma unroll
  for (int i = 0; i < 4; ++i) {
    const int off = 5 * i;
    const float p0 = cf[((long)n * 22 + off + 3) * chw + rem];
    const float p1 = cf[((long)n * 22 + off + 4) * chw + rem];
    const float loc0 = (float)u + (p0 - q0) * d;  // H coord
    const float loc1 = (float)v + (p1 - q1) * d;  // W coord
    const float g0 = (loc1 / (float)(W - 1) - 0.5f) * 2.0f;   // grid[...,0] (W)
    const float g1 = (loc0 / (float)(H - 1) - 0.5f) * 2.0f;   // grid[...,1] (H)
    const float x = ((g0 + 1.0f) * (float)W - 1.0f) * 0.5f;
    const float y = ((g1 + 1.0f) * (float)H - 1.0f) * 0.5f;
    const float x0f = floorf(x), y0f = floorf(y);
    float wx[4], wy[4];
    cubic_w(x - x0f, wx);
    cubic_w(y - y0f, wy);
    const int x0i = (int)x0f, y0i = (int)y0f;
    float a0 = 0.f, a1 = 0.f, a2 = 0.f;
#pragma unroll
    for (int ii = 0; ii < 4; ++ii) {
      const int yi = y0i - 1 + ii;
      const bool vy = (yi >= 0) && (yi < H);
      const int yic = min(max(yi, 0), H - 1);
#pragma unroll
      for (int jj = 0; jj < 4; ++jj) {
        const int xj = x0i - 1 + jj;
        const bool ok = vy && (xj >= 0) && (xj < W);
        const int xjc = min(max(xj, 0), W - 1);
        const float wv = ok ? wy[ii] * wx[jj] : 0.0f;
        const long base = ((long)n * 22 + off) * chw + (long)yic * W + xjc;
        a0 += wv * cf[base];
        a1 += wv * cf[base + chw];
        a2 += wv * cf[base + 2 * chw];
      }
    }
    out15[((long)n * 15 + 3 * i + 0) * chw + rem] = (bf16)a0;
    out15[((long)n * 15 + 3 * i + 1) * chw + rem] = (bf16)a1;
    out15[((long)n * 15 + 3 * i + 2) * chw + rem] = (bf16)a2;
  }
  out15[((long)n * 15 + 12) * chw + rem] = (bf16)d;
  out15[((long)n * 15 + 13) * chw + rem] = (bf16)q0;
  out15[((long)n * 15 + 14) * chw + rem] = (bf16)q1;
}

// ---------------------------------------------------------------------------
// Host orchestration
// ---------------------------------------------------------------------------
extern "C" void kernel_launch(void* const* d_in, const int* in_sizes, int n_in,
                              void* d_out, int out_size, void* d_ws, size_t ws_size,
                              hipStream_t stream) {
  (void)in_sizes; (void)n_in; (void)out_size; (void)ws_size;
  const float* dispF = (const float*)d_in[0];   // (4,200,268,268)
  const float* colF  = (const float*)d_in[1];   // (4,22,256,256)
  const float* dw1 = (const float*)d_in[2];  const float* db1 = (const float*)d_in[3];
  const float* dw2 = (const float*)d_in[4];  const float* db2 = (const float*)d_in[5];
  const float* dw3 = (const float*)d_in[6];  const float* db3 = (const float*)d_in[7];
  const float* dw4 = (const float*)d_in[8];  const float* db4 = (const float*)d_in[9];
  const float* cw1 = (const float*)d_in[10]; const float* cb1 = (const float*)d_in[11];
  const float* cw2 = (const float*)d_in[12]; const float* cb2 = (const float*)d_in[13];
  const float* cw3 = (const float*)d_in[14]; const float* cb3 = (const float*)d_in[15];
  const float* cw4 = (const float*)d_in[16]; const float* cb4 = (const float*)d_in[17];

  // ---- bump allocator over workspace ----
  char* ws = (char*)d_ws;
  size_t off = 0;
  auto alloc = [&](size_t bytes) -> void* {
    void* p = ws + off;
    off += (bytes + 255) & ~(size_t)255;
    return p;
  };
  auto padM = [](int oc) { return (oc + BM - 1) / BM * BM; };
  auto padK = [](int kt) { return (kt + BK - 1) / BK * BK; };

  const long szX  = 4L * 200 * 268 * 268;  // 57,459,200
  const long szR1 = 4L * 100 * 262 * 262;  // 27,457,600 (largest of A1/A3/C2)
  bf16* reg0 = (bf16*)alloc((size_t)szX * 2);   // X -> A2 -> C1 -> C3
  bf16* reg1 = (bf16*)alloc((size_t)szR1 * 2);  // A1 -> A3 -> C2
  bf16* W15  = (bf16*)alloc((size_t)(4L * 15 * 256 * 256) * 2);
  float* dsp = (float*)alloc((size_t)(4L * 256 * 256) * 4);

  struct WDesc { bf16* p; int OC, Ktot, Mpad, Kpad; };
  auto walloc = [&](int oc, int ktot) -> WDesc {
    WDesc d;
    d.OC = oc; d.Ktot = ktot; d.Mpad = padM(oc); d.Kpad = padK(ktot);
    d.p = (bf16*)alloc((size_t)d.Mpad * d.Kpad * 2);
    return d;
  };
  WDesc Wd1 = walloc(100, 200 * 49), Wd2 = walloc(100, 100 * 25);
  WDesc Wd3 = walloc(50, 100 * 9),   Wd4 = walloc(1, 50);
  WDesc Wc1 = walloc(100, 15 * 49),  Wc2 = walloc(100, 100 * 25);
  WDesc Wc3 = walloc(50, 100 * 9),   Wc4 = walloc(3, 50);

  auto cvt = [&](const float* src, bf16* dst, long n) {
    long b = (n + 255) / 256;
    int blocks = (int)(b > 4096 ? 4096 : b);
    f2bf_kernel<<<blocks, 256, 0, stream>>>(src, dst, n);
  };
  auto pack = [&](const float* src, const WDesc& d) {
    const long total = (long)d.Mpad * d.Kpad;
    long b = (total + 255) / 256;
    int blocks = (int)(b > 4096 ? 4096 : b);
    pack_wgt_kernel<<<blocks, 256, 0, stream>>>(src, d.p, d.OC, d.Ktot, d.Kpad, total);
  };
  auto conv = [&](const bf16* in, const WDesc& wd, const float* b, void* out,
                  int Nb, int C, int H, int W, int Kw, int relu, int obf) {
    const int OH = H - Kw + 1, OW = W - Kw + 1;
    const int Npix = Nb * OH * OW;
    dim3 grid((Npix + BN - 1) / BN, wd.Mpad / BM);
    switch (Kw) {
      case 7: conv_wmma_kernel<7><<<grid, 256, 0, stream>>>(in, wd.p, b, out, Nb, C, H, W, wd.OC, wd.Kpad, OH, OW, relu, obf); break;
      case 5: conv_wmma_kernel<5><<<grid, 256, 0, stream>>>(in, wd.p, b, out, Nb, C, H, W, wd.OC, wd.Kpad, OH, OW, relu, obf); break;
      case 3: conv_wmma_kernel<3><<<grid, 256, 0, stream>>>(in, wd.p, b, out, Nb, C, H, W, wd.OC, wd.Kpad, OH, OW, relu, obf); break;
      default: conv_wmma_kernel<1><<<grid, 256, 0, stream>>>(in, wd.p, b, out, Nb, C, H, W, wd.OC, wd.Kpad, OH, OW, relu, obf); break;
    }
  };

  // ---- conversions / weight packing ----
  cvt(dispF, reg0, szX);
  pack(dw1, Wd1); pack(dw2, Wd2); pack(dw3, Wd3); pack(dw4, Wd4);
  pack(cw1, Wc1); pack(cw2, Wc2); pack(cw3, Wc3); pack(cw4, Wc4);

  // ---- disparity net ----
  conv(reg0, Wd1, db1, reg1, 4, 200, 268, 268, 7, /*relu=*/1, /*obf=*/1); // A1
  conv(reg1, Wd2, db2, reg0, 4, 100, 262, 262, 5, 1, 1);                  // A2
  conv(reg0, Wd3, db3, reg1, 4, 100, 258, 258, 3, 1, 1);                  // A3
  conv(reg1, Wd4, db4, dsp,  4,  50, 256, 256, 1, 0, /*obf=*/0);          // disparity

  // ---- bicubic warps + channel assembly (15ch, bf16) ----
  {
    const int total = 4 * 256 * 256;
    warp_kernel<<<(total + 255) / 256, 256, 0, stream>>>(colF, dsp, W15, 4, 256, 256);
  }

  // ---- color net ----
  conv(W15,  Wc1, cb1, reg0, 4,  15, 256, 256, 7, 1, 1);                  // C1
  conv(reg0, Wc2, cb2, reg1, 4, 100, 250, 250, 5, 1, 1);                  // C2
  conv(reg1, Wc3, cb3, reg0, 4, 100, 246, 246, 3, 1, 1);                  // C3
  conv(reg0, Wc4, cb4, d_out, 4, 50, 244, 244, 1, 0, /*obf=*/0);          // final fp32
}
